// GaussianVectorQuantizer_74921409512024
// MI455X (gfx1250) — matrix operand order repaired and checked by reading, not compiled, not run on hardware
//
#include <hip/hip_runtime.h>
#include <hip/hip_bf16.h>
#include <math.h>

// ---- problem constants (from reference) ----
#define NB      64
#define NPTS    1024
#define DIM     64
#define BOOK    1024
#define NROWS   (NB * NPTS)          // 65536
#define ZQ_ELEMS ((size_t)NROWS * DIM)       // 4194304
#define LOGITS_ELEMS ((size_t)NROWS * BOOK)  // 67108864

typedef __attribute__((ext_vector_type(16))) __bf16 bf16x16;
typedef __attribute__((ext_vector_type(8)))  float  v8f;
typedef __attribute__((ext_vector_type(4)))  float  f4;

// Permuted book-fragment layout:
//   frag[((t*2 + s)*32 + lane)*16 + e]  (t = code tile 0..63, s = kstep 0..1)
//   c  = t*16 + (lane & 15)
//   k  = s*32 + (lane >> 4)*16 + e        (B fragment, contiguous k order, same as A)
__global__ __launch_bounds__(256) void gvq_prep_book_frags(
    const float* __restrict__ book, __bf16* __restrict__ fh, __bf16* __restrict__ fl)
{
    int tid = blockIdx.x * 256 + threadIdx.x;       // 65536 threads
    int e = tid & 15;
    int l = (tid >> 4) & 31;
    int s = (tid >> 9) & 1;
    int t = tid >> 10;
    int c = t * 16 + (l & 15);
    int k = s * 32 + (l >> 4) * 16 + e;
    float x = book[c * DIM + k];
    __bf16 hi = (__bf16)x;
    __bf16 lo = (__bf16)(x - (float)hi);
    fh[tid] = hi;
    fl[tid] = lo;
}

__global__ __launch_bounds__(256) void gvq_prep_misc(
    const float* __restrict__ book, const float* __restrict__ lpq,
    float* __restrict__ book2, float* __restrict__ prec_out)
{
    int c = blockIdx.x * 256 + threadIdx.x;         // 1024 threads
    if (c < BOOK) {
        float s = 0.f;
        #pragma unroll 8
        for (int k = 0; k < DIM; ++k) {
            float x = book[c * DIM + k];
            s += x * x;
        }
        book2[c] = s;
    }
    if (c == 0) {
        float pq = expf(lpq[0]);
        prec_out[0] = 0.5f / fmaxf(pq, 1e-10f);
    }
}

// Main kernel: 512 blocks x 256 threads (8 waves). Wave w of block b owns rows
// [b*128 + w*16, +16). It computes all 1024 logits for those rows via bf16x3
// WMMA, streams logits, tracks argmax, then gathers zq = book[argmax].
__global__ __launch_bounds__(256) void gvq_main(
    const float* __restrict__ ze, const float* __restrict__ book,
    const float* __restrict__ lpq,
    const __bf16* __restrict__ bfragHi, const __bf16* __restrict__ bfragLo,
    const float* __restrict__ book2,
    float* __restrict__ zq, float* __restrict__ logits)
{
    const int lane = threadIdx.x & 31;
    const int wave = threadIdx.x >> 5;
    const int rowBase = blockIdx.x * 128 + wave * 16;   // 16 rows per wave
    const int lrow = lane & 15;
    const int half = lane >> 4;                         // 0: lanes 0-15, 1: 16-31
    const int kb = half * 8;                            // unused; A-fragment K base is half*16

    const float precq = 0.5f / fmaxf(expf(lpq[0]), 1e-10f);

    // ---- Build A fragments (16x32 bf16, hi/lo split) for kstep 0 and 1 ----
    // A layout (16-bit, contiguous k, not the ISA table order): lane holds row M = lane&15;
//   elems 0..15 -> K = s*32 + half*16 + e (matches B fragment)
    bf16x16 aHi[2], aLo[2];
    float nrm = 0.f;
    const float* zrow = ze + (size_t)(rowBase + lrow) * DIM;
    #pragma unroll
    for (int s = 0; s < 2; ++s) {
        bf16x16 h, l;
        #pragma unroll
        for (int p = 0; p < 2; ++p) {                   // element halves
            int k = s * 32 + half * 16 + p * 8;
            f4 a = *(const f4*)(zrow + k);
            f4 b = *(const f4*)(zrow + k + 4);
            #pragma unroll
            for (int j = 0; j < 4; ++j) {
                float x = a[j];
                nrm += x * x;
                __bf16 hx = (__bf16)x;
                h[p * 8 + j] = hx;
                l[p * 8 + j] = (__bf16)(x - (float)hx);
                float y = b[j];
                nrm += y * y;
                __bf16 hy = (__bf16)y;
                h[p * 8 + 4 + j] = hy;
                l[p * 8 + 4 + j] = (__bf16)(y - (float)hy);
            }
        }
        aHi[s] = h;
        aLo[s] = l;
    }
    // lanes l and l+16 jointly hold row (l&15): combine partial norms
    nrm += __shfl_xor(nrm, 16);

    // Broadcast row norms: D element i corresponds to row (half*8 + i),
    // whose norm lives in lane (half*8 + i).
    float zn[8];
    #pragma unroll
    for (int i = 0; i < 8; ++i)
        zn[i] = __shfl(nrm, (half << 3) + i);

    float best[8];
    int   bidx[8];
    #pragma unroll
    for (int i = 0; i < 8; ++i) { best[i] = -3.4e38f; bidx[i] = 0; }

    const bf16x16* __restrict__ BH = (const bf16x16*)bfragHi;
    const bf16x16* __restrict__ BL = (const bf16x16*)bfragLo;

    // ---- Sweep all 64 code tiles ----
    for (int t = 0; t < 64; ++t) {
        const int c = t * 16 + lrow;                    // this lane's column
        bf16x16 bh0 = BH[(t * 2 + 0) * 32 + lane];
        bf16x16 bl0 = BL[(t * 2 + 0) * 32 + lane];
        bf16x16 bh1 = BH[(t * 2 + 1) * 32 + lane];
        bf16x16 bl1 = BL[(t * 2 + 1) * 32 + lane];

        v8f acc = {};
        // bf16x3: hi*hi + hi*lo + lo*hi  ~= fp32 product
        acc = __builtin_amdgcn_wmma_f32_16x16x32_bf16(false, aHi[0], false, bh0, (short)0, acc, false, false);
        acc = __builtin_amdgcn_wmma_f32_16x16x32_bf16(false, aHi[0], false, bl0, (short)0, acc, false, false);
        acc = __builtin_amdgcn_wmma_f32_16x16x32_bf16(false, aLo[0], false, bh0, (short)0, acc, false, false);
        acc = __builtin_amdgcn_wmma_f32_16x16x32_bf16(false, aHi[1], false, bh1, (short)0, acc, false, false);
        acc = __builtin_amdgcn_wmma_f32_16x16x32_bf16(false, aHi[1], false, bl1, (short)0, acc, false, false);
        acc = __builtin_amdgcn_wmma_f32_16x16x32_bf16(false, aLo[1], false, bh1, (short)0, acc, false, false);

        const float b2 = book2[c];
        #pragma unroll
        for (int i = 0; i < 8; ++i) {
            // D layout: VGPR i holds row (i + half*8), column (lane&15)
            float logit = -(zn[i] + b2 - 2.f * acc[i]) * precq;
            logits[(size_t)(rowBase + half * 8 + i) * BOOK + c] = logit;
            if (logit > best[i]) { best[i] = logit; bidx[i] = c; }
        }
    }

    // ---- argmax reduction across the 16 columns of each half-wave ----
    #pragma unroll
    for (int m = 8; m >= 1; m >>= 1) {
        #pragma unroll
        for (int i = 0; i < 8; ++i) {
            float ov = __shfl_xor(best[i], m);
            int   oi = __shfl_xor(bidx[i], m);
            if (ov > best[i] || (ov == best[i] && oi < bidx[i])) {
                best[i] = ov; bidx[i] = oi;
            }
        }
    }

    // ---- zq gather: row r's winner lives in element r&7 of half r>>3 ----
    #pragma unroll
    for (int r = 0; r < 16; ++r) {
        int bi = __shfl(bidx[r & 7], (r >> 3) << 4);    // lane 0 or lane 16
        float2 v = *(const float2*)(book + (size_t)bi * DIM + lane * 2);
        *(float2*)(zq + (size_t)(rowBase + r) * DIM + lane * 2) = v;
    }
}

extern "C" void kernel_launch(void* const* d_in, const int* in_sizes, int n_in,
                              void* d_out, int out_size, void* d_ws, size_t ws_size,
                              hipStream_t stream) {
    const float* ze   = (const float*)d_in[0];
    const float* book = (const float*)d_in[1];
    const float* lpq  = (const float*)d_in[2];
    (void)in_sizes; (void)n_in; (void)out_size; (void)ws_size;

    float* zq     = (float*)d_out;                 // [65536, 64]
    float* prec   = zq + ZQ_ELEMS;                 // [1]
    float* logits = prec + 1;                      // [65536, 1024]

    __bf16* fh    = (__bf16*)d_ws;                 // 64*2*32*16 = 65536 bf16 (128 KB)
    __bf16* fl    = fh + 64 * 2 * 32 * 16;         // 128 KB
    float*  book2 = (float*)(fl + 64 * 2 * 32 * 16); // 4 KB

    gvq_prep_book_frags<<<256, 256, 0, stream>>>(book, fh, fl);
    gvq_prep_misc<<<4, 256, 0, stream>>>(book, lpq, book2, prec);
    gvq_main<<<NROWS / 128, 256, 0, stream>>>(ze, book, lpq, fh, fl, book2, zq, logits);
}
